// TransformerDecoderLayer_22436909154460
// MI455X (gfx1250) — matrix operand order repaired and checked by reading, not compiled, not run on hardware
//
#include <hip/hip_runtime.h>
#include <stdint.h>

// ---------------------------------------------------------------------------
// Types for CDNA5 WMMA (wave32): bf16 operands, f32 accumulate.
// ---------------------------------------------------------------------------
typedef __bf16 bf16;
typedef __attribute__((ext_vector_type(16))) __bf16 v16bf;
typedef __attribute__((ext_vector_type(8)))  __bf16 v8bf;
typedef __attribute__((ext_vector_type(8)))  float  v8f;

union Frag16 { v16bf v; v8bf h[2]; };

__device__ __forceinline__ bf16 f2bf(float f) {
  union { float f; uint32_t u; } x; x.f = f;
  uint32_t r = x.u + 0x7FFFu + ((x.u >> 16) & 1u);   // round-to-nearest-even
  union { uint16_t s; bf16 b; } y; y.s = (uint16_t)(r >> 16);
  return y.b;
}

namespace cfg {
constexpr int T  = 1024, S = 2048, NB = 4, E = 1024, H = 16, FF = 4096;
constexpr int HD = E / H;     // 64
constexpr int BH = NB * H;    // 64 batch-heads
constexpr int MT = T * NB;    // 4096 target tokens
constexpr int MS = S * NB;    // 8192 memory tokens
constexpr int HC = 2;         // heads per attention chunk (workspace limit)
}

enum {
  EP_F32 = 0,        // outF[m*ldc+n] = v + bias
  EP_F32_RESID = 1,  // outF = v + bias + resid
  EP_BF16 = 2,       // outB = bf16(v + bias)
  EP_BF16_RELU = 3,  // outB = bf16(relu(v + bias))
  EP_HEAD_QK = 4,    // scatter to [head][token][hd] * scale (bf16)
  EP_HEAD_VT = 5,    // scatter transposed to [head][hd][token] (bf16)
  EP_ATTN_OUT = 6    // gather heads back to [token][E] (bf16)
};

// ---------------------------------------------------------------------------
// Generic WMMA GEMM: C[M,N] = A[M,K] (bf16) x B[N,K]^T (bf16) + epilogue.
// Block = 256 threads = 8 waves; wave tile = 16(M) x 64(N); block = 128x64.
// B operand: lane n%16 reads 16 contiguous K values of weight row n -> the
// natural PyTorch [N,K] weight layout needs no transpose.
// ---------------------------------------------------------------------------
__global__ __launch_bounds__(256)
void wmma_gemm_bf16(const bf16* __restrict__ A, int lda, long long sAz,
                    const bf16* __restrict__ B, int ldb, long long sBz,
                    const float* __restrict__ bias, int K, int emode,
                    float* __restrict__ outF, long long sOz,
                    bf16* __restrict__ outB,
                    const float* __restrict__ resid,
                    int ldc, float scale,
                    int NBp, int Hp, int HDp, int Ldim, int Ep, int hbase)
{
  const int lane = threadIdx.x & 31;
  const int wave = threadIdx.x >> 5;
  const int z    = blockIdx.z;
  const int m0   = blockIdx.y * 128 + wave * 16;
  const int n0   = blockIdx.x * 64;

  const bf16* Ab = A + (long long)z * sAz;
  const bf16* Bb = B + (long long)z * sBz;

  const int l15  = lane & 15;
  const int aoff = (lane < 16) ? 0 : 16;  // A: K half per lane-half
  const int boff = (lane < 16) ? 0 : 16;  // B: K half per lane-half

  const bf16* Arow  = Ab + (long long)(m0 + l15) * lda + aoff;
  const bf16* Brow0 = Bb + (long long)(n0 +  0 + l15) * ldb + boff;
  const bf16* Brow1 = Bb + (long long)(n0 + 16 + l15) * ldb + boff;
  const bf16* Brow2 = Bb + (long long)(n0 + 32 + l15) * ldb + boff;
  const bf16* Brow3 = Bb + (long long)(n0 + 48 + l15) * ldb + boff;

  v8f acc0 = {}, acc1 = {}, acc2 = {}, acc3 = {};

  for (int kb = 0; kb < K; kb += 32) {
    __builtin_prefetch(Arow + kb + 128, 0, 1);   // global_prefetch_b8
    __builtin_prefetch(Brow0 + kb + 128, 0, 1);

    Frag16 a;
    a.v = *(const v16bf*)(Arow + kb);            // K = kb+aoff .. +15
    Frag16 b0, b1, b2, b3;
    b0.v = *(const v16bf*)(Brow0 + kb);          // 16 contiguous K
    b1.v = *(const v16bf*)(Brow1 + kb);
    b2.v = *(const v16bf*)(Brow2 + kb);
    b3.v = *(const v16bf*)(Brow3 + kb);

    acc0 = __builtin_amdgcn_wmma_f32_16x16x32_bf16(false, a.v, false, b0.v, (short)0, acc0, false, false);
    acc1 = __builtin_amdgcn_wmma_f32_16x16x32_bf16(false, a.v, false, b1.v, (short)0, acc1, false, false);
    acc2 = __builtin_amdgcn_wmma_f32_16x16x32_bf16(false, a.v, false, b2.v, (short)0, acc2, false, false);
    acc3 = __builtin_amdgcn_wmma_f32_16x16x32_bf16(false, a.v, false, b3.v, (short)0, acc3, false, false);
  }

  // C/D layout: lanes 0-15 -> M = m0+r, lanes 16-31 -> M = m0+8+r; N = n0+j*16+l15
  const int mb = m0 + ((lane < 16) ? 0 : 8);
  float* Of = outF ? (outF + (long long)z * sOz) : nullptr;
  v8f accs[4] = {acc0, acc1, acc2, acc3};

#pragma unroll
  for (int j = 0; j < 4; ++j) {
    const int n = n0 + j * 16 + l15;
    const float bs = bias ? bias[n] : 0.0f;
#pragma unroll
    for (int r = 0; r < 8; ++r) {
      const int m = mb + r;
      const float v = accs[j][r] + bs;
      const long long idx = (long long)m * ldc + n;
      switch (emode) {
        case EP_F32:       Of[idx] = v; break;
        case EP_F32_RESID: Of[idx] = v + resid[idx]; break;
        case EP_BF16:      outB[idx] = f2bf(v); break;
        case EP_BF16_RELU: outB[idx] = f2bf(v > 0.0f ? v : 0.0f); break;
        case EP_HEAD_QK: {  // token rows are (t, nb) interleaved; cols are (h, d)
          const int t = m / NBp, nb = m % NBp, h = n / HDp, d = n % HDp;
          const int b = nb * Hp + h;
          outB[((long long)b * Ldim + t) * HDp + d] = f2bf(v * scale);
        } break;
        case EP_HEAD_VT: {  // V stored transposed per head: [b][d][token]
          const int t = m / NBp, nb = m % NBp, h = n / HDp, d = n % HDp;
          const int b = nb * Hp + h;
          outB[((long long)b * HDp + d) * Ldim + t] = f2bf(v);
        } break;
        case EP_ATTN_OUT: { // m = t, n = d, head = hbase + z
          const int b = hbase + z, nb = b / Hp, h = b % Hp;
          outB[((long long)m * NBp + nb) * Ep + h * HDp + n] = f2bf(v);
        } break;
      }
    }
  }
}

// ---------------------------------------------------------------------------
// Row softmax: f32 scores -> bf16 probabilities. One 256-thread block per row.
// ---------------------------------------------------------------------------
__global__ __launch_bounds__(256)
void softmax_rows(const float* __restrict__ Sm, bf16* __restrict__ P, int cols)
{
  __shared__ float red[256];
  const int row = blockIdx.x;
  const int tid = threadIdx.x;
  const float* x = Sm + (long long)row * cols;
  bf16* y = P + (long long)row * cols;

  float mx = -3.4e38f;
  for (int c = tid; c < cols; c += 256) mx = fmaxf(mx, x[c]);
  red[tid] = mx; __syncthreads();
  for (int s = 128; s > 0; s >>= 1) { if (tid < s) red[tid] = fmaxf(red[tid], red[tid + s]); __syncthreads(); }
  mx = red[0]; __syncthreads();

  float sum = 0.0f;
  for (int c = tid; c < cols; c += 256) sum += __expf(x[c] - mx);
  red[tid] = sum; __syncthreads();
  for (int s = 128; s > 0; s >>= 1) { if (tid < s) red[tid] += red[tid + s]; __syncthreads(); }
  const float inv = 1.0f / red[0];

  for (int c = tid; c < cols; c += 256) y[c] = f2bf(__expf(x[c] - mx) * inv);
}

// ---------------------------------------------------------------------------
// LayerNorm over last dim; emits f32 (residual chain) and optional bf16 copy.
// ---------------------------------------------------------------------------
__global__ __launch_bounds__(256)
void layernorm_rows(const float* __restrict__ X, const float* __restrict__ g,
                    const float* __restrict__ bta, float* __restrict__ Yf,
                    bf16* __restrict__ Yb, int cols)
{
  __shared__ float red[256];
  __shared__ float mu_s, rstd_s;
  const int row = blockIdx.x;
  const int tid = threadIdx.x;
  const float* x = X + (long long)row * cols;

  float s = 0.0f;
  for (int c = tid; c < cols; c += 256) s += x[c];
  red[tid] = s; __syncthreads();
  for (int k = 128; k > 0; k >>= 1) { if (tid < k) red[tid] += red[tid + k]; __syncthreads(); }
  if (tid == 0) mu_s = red[0] / (float)cols;
  __syncthreads();
  const float mu = mu_s;

  float q = 0.0f;
  for (int c = tid; c < cols; c += 256) { const float d = x[c] - mu; q += d * d; }
  red[tid] = q; __syncthreads();
  for (int k = 128; k > 0; k >>= 1) { if (tid < k) red[tid] += red[tid + k]; __syncthreads(); }
  if (tid == 0) rstd_s = rsqrtf(red[0] / (float)cols + 1e-5f);
  __syncthreads();
  const float rstd = rstd_s;

  for (int c = tid; c < cols; c += 256) {
    const float v = (x[c] - mu) * rstd * g[c] + bta[c];
    Yf[(long long)row * cols + c] = v;
    if (Yb) Yb[(long long)row * cols + c] = f2bf(v);
  }
}

__global__ void cast_f32_to_bf16(const float* __restrict__ x, bf16* __restrict__ y, long long n)
{
  long long i = (long long)blockIdx.x * blockDim.x + threadIdx.x;
  const long long stride = (long long)gridDim.x * blockDim.x;
  for (; i < n; i += stride) y[i] = f2bf(x[i]);
}

// ---------------------------------------------------------------------------
// Host-side orchestration (graph-capture safe: only kernel launches).
// ---------------------------------------------------------------------------
extern "C" void kernel_launch(void* const* d_in, const int* in_sizes, int n_in,
                              void* d_out, int out_size, void* d_ws, size_t ws_size,
                              hipStream_t stream) {
  using namespace cfg;
  (void)in_sizes; (void)n_in; (void)out_size; (void)ws_size;

  const float* tgt      = (const float*)d_in[0];
  const float* memory   = (const float*)d_in[1];
  const float* sa_w_in  = (const float*)d_in[2];
  const float* sa_b_in  = (const float*)d_in[3];
  const float* sa_w_out = (const float*)d_in[4];
  const float* sa_b_out = (const float*)d_in[5];
  const float* ca_w_in  = (const float*)d_in[6];
  const float* ca_b_in  = (const float*)d_in[7];
  const float* ca_w_out = (const float*)d_in[8];
  const float* ca_b_out = (const float*)d_in[9];
  const float* w1       = (const float*)d_in[10];
  const float* b1       = (const float*)d_in[11];
  const float* w2       = (const float*)d_in[12];
  const float* b2       = (const float*)d_in[13];
  const float* ln1_g    = (const float*)d_in[14];
  const float* ln1_b    = (const float*)d_in[15];
  const float* ln2_g    = (const float*)d_in[16];
  const float* ln2_b    = (const float*)d_in[17];
  const float* ln3_g    = (const float*)d_in[18];
  const float* ln3_b    = (const float*)d_in[19];
  float* out = (float*)d_out;

  uintptr_t wptr = (uintptr_t)d_ws;
  auto alloc = [&](size_t bytes) -> void* {
    uintptr_t p = (wptr + 255) & ~(uintptr_t)255;
    wptr = p + bytes;
    return (void*)p;
  };

  bf16* tgt_b    = (bf16*)alloc((size_t)MT * E * 2);
  bf16* mem_b    = (bf16*)alloc((size_t)MS * E * 2);
  bf16* sa_win_b = (bf16*)alloc((size_t)3 * E * E * 2);
  bf16* sa_wout_b= (bf16*)alloc((size_t)E * E * 2);
  bf16* ca_win_b = (bf16*)alloc((size_t)3 * E * E * 2);
  bf16* ca_wout_b= (bf16*)alloc((size_t)E * E * 2);
  bf16* w1_b     = (bf16*)alloc((size_t)FF * E * 2);
  bf16* w2_b     = (bf16*)alloc((size_t)E * FF * 2);
  bf16* q_b      = (bf16*)alloc((size_t)BH * T * HD * 2);
  bf16* k_sa     = (bf16*)alloc((size_t)BH * T * HD * 2);
  bf16* v_sa     = (bf16*)alloc((size_t)BH * HD * T * 2);
  bf16* k_ca     = (bf16*)alloc((size_t)BH * S * HD * 2);
  bf16* v_ca     = (bf16*)alloc((size_t)BH * HD * S * 2);
  bf16* attn_b   = (bf16*)alloc((size_t)MT * E * 2);
  float* pre     = (float*)alloc((size_t)MT * E * 4);
  float* x1f     = (float*)alloc((size_t)MT * E * 4);
  bf16* x1b      = (bf16*)alloc((size_t)MT * E * 2);
  float* x2f     = (float*)alloc((size_t)MT * E * 4);
  bf16* x2b      = (bf16*)alloc((size_t)MT * E * 2);
  bf16* h1_b     = (bf16*)alloc((size_t)MT * FF * 2);
  float* scores  = (float*)alloc((size_t)HC * T * S * 4);
  bf16* probs    = (bf16*)alloc((size_t)HC * T * S * 2);

  auto cast = [&](const float* src, bf16* dst, long long n) {
    cast_f32_to_bf16<<<1024, 256, 0, stream>>>(src, dst, n);
  };
  cast(tgt, tgt_b, (long long)MT * E);
  cast(memory, mem_b, (long long)MS * E);
  cast(sa_w_in, sa_win_b, 3LL * E * E);
  cast(sa_w_out, sa_wout_b, (long long)E * E);
  cast(ca_w_in, ca_win_b, 3LL * E * E);
  cast(ca_w_out, ca_wout_b, (long long)E * E);
  cast(w1, w1_b, (long long)FF * E);
  cast(w2, w2_b, (long long)E * FF);

  auto gemm = [&](const bf16* A, int lda, long long sAz,
                  const bf16* B, int ldb, long long sBz,
                  const float* bias, int M, int Nn, int K, int Z, int emode,
                  float* outF, long long sOz, bf16* outB, const float* resid,
                  int ldc, float scale, int Ldim, int hbase) {
    dim3 grid(Nn / 64, M / 128, Z);
    wmma_gemm_bf16<<<grid, 256, 0, stream>>>(A, lda, sAz, B, ldb, sBz, bias, K,
        emode, outF, sOz, outB, resid, ldc, scale, NB, H, HD, Ldim, E, hbase);
  };

  auto attention = [&](const bf16* q, const bf16* k, const bf16* vt, int Sl) {
    for (int hc = 0; hc < BH; hc += HC) {
      // scores[z][t][s] = q_head . k_head  (K = HD)
      gemm(q + (long long)hc * T * HD, HD, (long long)T * HD,
           k + (long long)hc * Sl * HD, HD, (long long)Sl * HD,
           nullptr, T, Sl, HD, HC, EP_F32,
           scores, (long long)T * Sl, nullptr, nullptr, Sl, 1.0f, 0, 0);
      softmax_rows<<<HC * T, 256, 0, stream>>>(scores, probs, Sl);
      // O[t][d] = P . V  (K = Sl, V stored transposed -> contiguous B operand)
      gemm(probs, Sl, (long long)T * Sl,
           vt + (long long)hc * HD * Sl, Sl, (long long)HD * Sl,
           nullptr, T, HD, Sl, HC, EP_ATTN_OUT,
           nullptr, 0, attn_b, nullptr, 0, 1.0f, 0, hc);
    }
  };

  const float qscale = 0.125f;  // HD^-0.5 with HD = 64

  // ---- self-attention ----
  gemm(tgt_b, E, 0, sa_win_b,                    E, 0, sa_b_in,         MT, E, E, 1, EP_HEAD_QK, nullptr, 0, q_b,  nullptr, 0, qscale, T, 0);
  gemm(tgt_b, E, 0, sa_win_b + (size_t)E * E,    E, 0, sa_b_in + E,     MT, E, E, 1, EP_HEAD_QK, nullptr, 0, k_sa, nullptr, 0, 1.0f,   T, 0);
  gemm(tgt_b, E, 0, sa_win_b + (size_t)2 * E * E,E, 0, sa_b_in + 2 * E, MT, E, E, 1, EP_HEAD_VT, nullptr, 0, v_sa, nullptr, 0, 1.0f,   T, 0);
  attention(q_b, k_sa, v_sa, T);
  gemm(attn_b, E, 0, sa_wout_b, E, 0, sa_b_out, MT, E, E, 1, EP_F32_RESID, pre, 0, nullptr, tgt, E, 1.0f, 0, 0);
  layernorm_rows<<<MT, 256, 0, stream>>>(pre, ln1_g, ln1_b, x1f, x1b, E);

  // ---- cross-attention ----
  gemm(x1b,   E, 0, ca_win_b,                    E, 0, ca_b_in,         MT, E, E, 1, EP_HEAD_QK, nullptr, 0, q_b,  nullptr, 0, qscale, T, 0);
  gemm(mem_b, E, 0, ca_win_b + (size_t)E * E,    E, 0, ca_b_in + E,     MS, E, E, 1, EP_HEAD_QK, nullptr, 0, k_ca, nullptr, 0, 1.0f,   S, 0);
  gemm(mem_b, E, 0, ca_win_b + (size_t)2 * E * E,E, 0, ca_b_in + 2 * E, MS, E, E, 1, EP_HEAD_VT, nullptr, 0, v_ca, nullptr, 0, 1.0f,   S, 0);
  attention(q_b, k_ca, v_ca, S);
  gemm(attn_b, E, 0, ca_wout_b, E, 0, ca_b_out, MT, E, E, 1, EP_F32_RESID, pre, 0, nullptr, x1f, E, 1.0f, 0, 0);
  layernorm_rows<<<MT, 256, 0, stream>>>(pre, ln2_g, ln2_b, x2f, x2b, E);

  // ---- FFN ----
  gemm(x2b,  E,  0, w1_b, E,  0, b1, MT, FF, E,  1, EP_BF16_RELU, nullptr, 0, h1_b,    nullptr, FF, 1.0f, 0, 0);
  gemm(h1_b, FF, 0, w2_b, FF, 0, b2, MT, E,  FF, 1, EP_F32_RESID, pre,     0, nullptr, x2f,     E,  1.0f, 0, 0);
  layernorm_rows<<<MT, 256, 0, stream>>>(pre, ln3_g, ln3_b, out, nullptr, E);
}
